// ODE_RNN_50328426774884
// MI455X (gfx1250) — matrix operand order repaired and checked by reading, not compiled, not run on hardware
//
#include <hip/hip_runtime.h>
#include <math.h>

// ---------------- problem constants (match reference) ----------------
constexpr int HID    = 2048;
constexpr int NIN    = 512;
constexpr int NOUT   = 512;

constexpr int NWG    = 32;            // persistent workgroups
constexpr int TPB    = 128;           // 4 waves of 32
constexpr int WPG    = TPB / 32;      // waves per WG
constexpr int DTILES = NOUT / 16;     // 32 decoder tiles
constexpr int KCH_H  = HID / 32;      // 64 k-chunks for K=2048
constexpr int KCH_I  = NIN / 32;      // 16 k-chunks for K=512
constexpr int ZPAD_N = HID + 32;      // full-size zero buffer (halves)

typedef __attribute__((ext_vector_type(16))) __bf16 v16bf;
typedef __attribute__((ext_vector_type(8)))  float  v8f;

// ---------------- helpers ----------------
__device__ __forceinline__ unsigned f32_to_bf16_bits(float f) {
    unsigned u = __float_as_uint(f);
    u += 0x7FFFu + ((u >> 16) & 1u);        // round-to-nearest-even
    return u >> 16;
}

// Pack a row-major f32 weight [rows x cols] into bf16 WMMA A-fragment order:
// (row,k) -> tile t=row/16, chunk c=k/32; lane = (row%16) + 16*((k%32)/16);
// j = k%16. Each lane's 16 halves are contiguous (32B load).
__global__ void pack_bf16_kernel(const float* __restrict__ W,
                                 unsigned short* __restrict__ P,
                                 int rows, int cols) {
    int idx = blockIdx.x * blockDim.x + threadIdx.x;
    if (idx >= rows * cols) return;
    int row = idx / cols, k = idx % cols;
    int t = row >> 4, lr = row & 15;
    int c = k >> 5,  kk = k & 31;
    int grp = kk >> 3;
    int lane = lr + ((grp >> 1) << 4);
    int j = kk & 15;
    int kch = cols >> 5;
    size_t dst = (((size_t)t * kch + c) * 32 + lane) * 16 + j;
    P[dst] = (unsigned short)f32_to_bf16_bits(W[idx]);
}

// Plain elementwise f32 -> bf16 (for the x sequence).
__global__ void cvt_bf16_kernel(const float* __restrict__ src,
                                unsigned short* __restrict__ dst, int n) {
    int idx = blockIdx.x * blockDim.x + threadIdx.x;
    if (idx < n) dst[idx] = (unsigned short)f32_to_bf16_bits(src[idx]);
}

__global__ void init_misc_kernel(unsigned* bar, unsigned short* zpad) {
    int idx = blockIdx.x * blockDim.x + threadIdx.x;
    if (idx == 0) *bar = 0u;
    if (idx < ZPAD_N) zpad[idx] = 0;
}

// Pack 8 floats into 8 bf16 and store as one 16B write.
__device__ __forceinline__ void store_bf16x8(unsigned short* dst, const float* v) {
    uint4 u;
    u.x = f32_to_bf16_bits(v[0]) | (f32_to_bf16_bits(v[1]) << 16);
    u.y = f32_to_bf16_bits(v[2]) | (f32_to_bf16_bits(v[3]) << 16);
    u.z = f32_to_bf16_bits(v[4]) | (f32_to_bf16_bits(v[5]) << 16);
    u.w = f32_to_bf16_bits(v[6]) | (f32_to_bf16_bits(v[7]) << 16);
    *reinterpret_cast<uint4*>(dst) = u;
}

// y(16 rows of `tile`) += Wpacked[tile,:] * vec   via chained bf16 WMMAs.
// Branch-free B fragments: lanes 0/16 stream the live vector (B column 0);
// all other lanes walk a full-size zero buffer at the SAME stride, so every
// lane uses one base pointer + immediate offsets (no per-chunk address adds).
__device__ __forceinline__ v8f mv_tile(const unsigned short* __restrict__ P,
                                       const unsigned short* __restrict__ vec,
                                       const unsigned short* __restrict__ zpad,
                                       int tile, int kch, int lane, v8f cin) {
    const v16bf* ap = reinterpret_cast<const v16bf*>(P)
                      + (size_t)tile * kch * 32 + lane;
    const bool bl = (lane == 0) || (lane == 16);
    const unsigned short* bp = (bl ? vec : zpad) + ((lane == 16) ? 16 : 0);
    v8f c0 = cin; v8f c1 = {}; v8f c2 = {}; v8f c3 = {};
    for (int c = 0; c < kch; c += 4) {
        v16bf a0 = ap[(size_t)(c + 0) * 32];
        v16bf a1 = ap[(size_t)(c + 1) * 32];
        v16bf a2 = ap[(size_t)(c + 2) * 32];
        v16bf a3 = ap[(size_t)(c + 3) * 32];
        v16bf b0 = *reinterpret_cast<const v16bf*>(bp + (c + 0) * 32);
        v16bf b1 = *reinterpret_cast<const v16bf*>(bp + (c + 1) * 32);
        v16bf b2 = *reinterpret_cast<const v16bf*>(bp + (c + 2) * 32);
        v16bf b3 = *reinterpret_cast<const v16bf*>(bp + (c + 3) * 32);
        c0 = __builtin_amdgcn_wmma_f32_16x16x32_bf16(false, a0, false, b0, (short)0, c0, false, false);
        c1 = __builtin_amdgcn_wmma_f32_16x16x32_bf16(false, a1, false, b1, (short)0, c1, false, false);
        c2 = __builtin_amdgcn_wmma_f32_16x16x32_bf16(false, a2, false, b2, (short)0, c2, false, false);
        c3 = __builtin_amdgcn_wmma_f32_16x16x32_bf16(false, a3, false, b3, (short)0, c3, false, false);
    }
    return (c0 + c1) + (c2 + c3);
}

// ---------------- persistent ODE-RNN kernel ----------------
__global__ __launch_bounds__(TPB) void ode_rnn_kernel(
    const unsigned short* __restrict__ xbf, const float* __restrict__ tvec, int T,
    const unsigned short* __restrict__ pWin,  const float* __restrict__ b_in,
    const unsigned short* __restrict__ pWhid, const float* __restrict__ b_hid,
    const unsigned short* __restrict__ pWode, const float* __restrict__ b_ode,
    const unsigned short* __restrict__ pWdec, const float* __restrict__ b_dec,
    const int* __restrict__ step_ptr,
    unsigned short* hbfA, unsigned short* hbfB,
    unsigned short* zbfA, unsigned short* zbfB,
    const unsigned short* __restrict__ zpad,
    unsigned* bar, float* __restrict__ out)
{
    const int tid  = threadIdx.x;
    const int lane = tid & 31;
    const int wave = tid >> 5;
    const int tile = blockIdx.x * WPG + wave;           // 0..127
    const int nsteps = *step_ptr;
    // row base this lane owns in the D fragment (col 0): lane0 -> M0-7, lane16 -> M8-15
    const int m0 = (lane == 0) ? tile * 16 : ((lane == 16) ? tile * 16 + 8 : -1);

    // lane-private f32 state for the 8 owned rows (registers, never spilled)
    float rh[8];                       // h (f32 master copy)
    float racc[8];                     // RK4 accumulator k1+2k2+2k3
    float rbode[8], rbinh[8];          // biases
    if (m0 >= 0) {
        #pragma unroll
        for (int r = 0; r < 8; ++r) {
            rbode[r] = b_ode[m0 + r];
            rbinh[r] = b_in[m0 + r] + b_hid[m0 + r];
        }
    }

    unsigned epoch = 0;
    auto gbar = [&]() {   // device-wide barrier over NWG workgroups
        __threadfence();
        __syncthreads();
        ++epoch;
        if (tid == 0) {
            __hip_atomic_fetch_add(bar, 1u, __ATOMIC_RELEASE, __HIP_MEMORY_SCOPE_AGENT);
            const unsigned tgt = (unsigned)NWG * epoch;
            while (__hip_atomic_load(bar, __ATOMIC_ACQUIRE, __HIP_MEMORY_SCOPE_AGENT) < tgt)
                __builtin_amdgcn_s_sleep(2);
        }
        __syncthreads();
        __threadfence();
    };

    unsigned short* hc = hbfA;   // current h (bf16, published)
    unsigned short* hn = hbfB;

    // ---- init: h0 = tanh(W_in x0 + b_in + b_hid)
    {
        v8f cf = {};
        cf = mv_tile(pWin, xbf, zpad, tile, KCH_I, lane, cf);
        if (m0 >= 0) {
            float zv[8];
            #pragma unroll
            for (int r = 0; r < 8; ++r) { rh[r] = tanhf(cf[r] + rbinh[r]); zv[r] = rh[r]; }
            store_bf16x8(hc + m0, zv);
        }
        gbar();
    }

    for (int i = 1; i < T; ++i) {
        const float dt = (tvec[i] - tvec[i - 1]) / (float)nsteps;
        for (int s = 0; s < nsteps; ++s) {
            #pragma unroll 1
            for (int ph = 0; ph < 4; ++ph) {
                // k1: h->zA, k2: zA->zB, k3: zB->zA, k4: zA->h (bf16 publish)
                const unsigned short* zin = (ph == 0) ? hc : ((ph == 2) ? zbfB : zbfA);
                unsigned short* zout = (ph == 3) ? hc : ((ph == 1) ? zbfB : zbfA);
                v8f cf = {};
                cf = mv_tile(pWode, zin, zpad, tile, KCH_H, lane, cf);
                if (m0 >= 0) {
                    float zv[8];
                    #pragma unroll
                    for (int r = 0; r < 8; ++r) {
                        float wv = tanhf(cf[r] + rbode[r]);
                        if (ph == 0)      { racc[r]  = wv;        zv[r] = rh[r] + 0.5f * dt * wv; }
                        else if (ph == 1) { racc[r] += 2.f * wv;  zv[r] = rh[r] + 0.5f * dt * wv; }
                        else if (ph == 2) { racc[r] += 2.f * wv;  zv[r] = rh[r] + dt * wv; }
                        else              { rh[r]   += (dt / 6.f) * (racc[r] + wv); zv[r] = rh[r]; }
                    }
                    store_bf16x8(zout + m0, zv);
                }
                gbar();
            }
        }
        // RNN update: h = tanh(W_in x_i + W_hid h + b_in + b_hid) -> publish to hn
        {
            v8f cf = {};
            cf = mv_tile(pWin,  xbf + (size_t)i * NIN, zpad, tile, KCH_I, lane, cf);
            cf = mv_tile(pWhid, hc,                    zpad, tile, KCH_H, lane, cf);
            if (m0 >= 0) {
                float zv[8];
                #pragma unroll
                for (int r = 0; r < 8; ++r) { rh[r] = tanhf(cf[r] + rbinh[r]); zv[r] = rh[r]; }
                store_bf16x8(hn + m0, zv);
            }
            gbar();
        }
        unsigned short* tsw = hc; hc = hn; hn = tsw;
    }

    // ---- decode: out = W_dec h + b_dec (tiles 0..31; wave-uniform guard)
    if (tile < DTILES) {
        v8f cf = {};
        cf = mv_tile(pWdec, hc, zpad, tile, KCH_H, lane, cf);
        if (m0 >= 0) {
            #pragma unroll
            for (int r = 0; r < 8; ++r) {
                int m = m0 + r;
                out[m] = cf[r] + b_dec[m];
            }
        }
    }
}

// ---------------- host entry ----------------
extern "C" void kernel_launch(void* const* d_in, const int* in_sizes, int n_in,
                              void* d_out, int out_size, void* d_ws, size_t ws_size,
                              hipStream_t stream)
{
    const float* x     = (const float*)d_in[0];
    const float* tvec  = (const float*)d_in[1];
    const float* W_in  = (const float*)d_in[2];
    const float* b_in  = (const float*)d_in[3];
    const float* W_hid = (const float*)d_in[4];
    const float* b_hid = (const float*)d_in[5];
    const float* W_ode = (const float*)d_in[6];
    const float* b_ode = (const float*)d_in[7];
    const float* W_dec = (const float*)d_in[8];
    const float* b_dec = (const float*)d_in[9];
    const int*   stepp = (const int*)d_in[10];
    float* out = (float*)d_out;
    const int T = in_sizes[1];

    char* ws = (char*)d_ws;
    unsigned short* pWode = (unsigned short*)(ws);                              // 8 MB
    unsigned short* pWhid = (unsigned short*)(ws + (size_t)8  * 1024 * 1024);   // 8 MB
    unsigned short* pWin  = (unsigned short*)(ws + (size_t)16 * 1024 * 1024);   // 2 MB
    unsigned short* pWdec = (unsigned short*)(ws + (size_t)18 * 1024 * 1024);   // 2 MB
    unsigned short* xbf   = (unsigned short*)(ws + (size_t)20 * 1024 * 1024);   // T*NIN*2 = 1 MB
    char* vecbase = ws + (size_t)21 * 1024 * 1024 + 512 * 1024;
    unsigned short* hbfA = (unsigned short*)(vecbase);
    unsigned short* hbfB = hbfA + HID;
    unsigned short* zbfA = hbfB + HID;
    unsigned short* zbfB = zbfA + HID;
    unsigned short* zpad = zbfB + HID;            // ZPAD_N zero halves
    unsigned* bar = (unsigned*)(zpad + ZPAD_N);

    int n;
    n = HID * HID;
    pack_bf16_kernel<<<(n + 255) / 256, 256, 0, stream>>>(W_ode, pWode, HID, HID);
    pack_bf16_kernel<<<(n + 255) / 256, 256, 0, stream>>>(W_hid, pWhid, HID, HID);
    n = HID * NIN;
    pack_bf16_kernel<<<(n + 255) / 256, 256, 0, stream>>>(W_in,  pWin,  HID, NIN);
    n = NOUT * HID;
    pack_bf16_kernel<<<(n + 255) / 256, 256, 0, stream>>>(W_dec, pWdec, NOUT, HID);
    n = T * NIN;
    cvt_bf16_kernel<<<(n + 255) / 256, 256, 0, stream>>>(x, xbf, n);
    init_misc_kernel<<<(ZPAD_N + 255) / 256, 256, 0, stream>>>(bar, zpad);

    ode_rnn_kernel<<<NWG, TPB, 0, stream>>>(
        xbf, tvec, T, pWin, b_in, pWhid, b_hid, pWode, b_ode, pWdec, b_dec,
        stepp, hbfA, hbfB, zbfA, zbfB, zpad, bar, out);
}
